// GraphSAGEConv_25237227831414
// MI455X (gfx1250) — matrix operand order repaired and checked by reading, not compiled, not run on hardware
//
#include <hip/hip_runtime.h>
#include <hip/hip_bf16.h>
#include <cstdint>
#include <cstddef>

// ---- problem constants (match reference) ----
#define RANK    512
#define NPG     9
#define E_PER   12
#define NGRAPH  16384
#define NNODES  (NGRAPH * NPG)   // 147456
#define LN_EPS  1e-5f

typedef __attribute__((ext_vector_type(16))) __bf16 v16bf;
typedef __attribute__((ext_vector_type(8)))  float  v8f;
typedef __attribute__((ext_vector_type(4)))  unsigned int v4u;
typedef __attribute__((ext_vector_type(8)))  int    v8i;
typedef __attribute__((ext_vector_type(4)))  int    v4i;

struct Frag32 { uint4 q0, q1; };

static __device__ __forceinline__ v16bf make_frag(uint4 a, uint4 b) {
  Frag32 f; f.q0 = a; f.q1 = b;
  return __builtin_bit_cast(v16bf, f);
}

// fp32 -> bf16 bits, round-to-nearest-even
static __device__ __forceinline__ unsigned short bf16_bits(float f) {
  unsigned int u = __builtin_bit_cast(unsigned int, f);
  u += 0x7fffu + ((u >> 16) & 1u);
  return (unsigned short)(u >> 16);
}
static __device__ __forceinline__ float bf16_f32(unsigned short b) {
  return __builtin_bit_cast(float, ((unsigned int)b) << 16);
}

// ---------------------------------------------------------------------------
// Pre-pass: W[l][k][n] fp32 -> WT[l][n][k] bf16 (transposed so a WMMA
// B-fragment = two runs of 8 consecutive K at fixed N = two 16-byte loads).
// ---------------------------------------------------------------------------
__global__ void k_convert_w(const float* __restrict__ Ws,
                            unsigned short* __restrict__ WT) {
  int idx = blockIdx.x * 256 + threadIdx.x;   // 0 .. 3*512*512-1
  int l = idx >> 18;
  int r = idx & 262143;
  int k = r >> 9;
  int n = r & 511;
  WT[l * 262144 + n * 512 + k] = bf16_bits(Ws[idx]);
}

// ---------------------------------------------------------------------------
// Per-graph aggregation: degrees + norms from the 12 local edges, scatter-add
// in registers (thread owns feature columns t and t+256), emit m*in_norm bf16.
// ---------------------------------------------------------------------------
template <bool FIRST>
__global__ void k_agg(const int* __restrict__ feats,
                      const int* __restrict__ src,
                      const int* __restrict__ dst,
                      const float* __restrict__ emb,
                      const unsigned short* __restrict__ h_prev,
                      unsigned short* __restrict__ a_out) {
  __shared__ int   ls[E_PER], ld[E_PER];
  __shared__ int   odeg[NPG], ideg[NPG];
  __shared__ float onorm[NPG], inorm[NPG];

  const int g = blockIdx.x;
  const int t = threadIdx.x;

  if (t < NPG) { odeg[t] = 0; ideg[t] = 0; }
  __syncthreads();
  if (t < E_PER) {
    int s = src[g * E_PER + t] - g * NPG;   // local ids in [0, NPG)
    int d = dst[g * E_PER + t] - g * NPG;
    ls[t] = s; ld[t] = d;
    atomicAdd(&odeg[s], 1);
    atomicAdd(&ideg[d], 1);
  }
  __syncthreads();
  if (t < NPG) {
    onorm[t] = rsqrtf(odeg[t] > 0 ? (float)odeg[t] : 1.0f);
    inorm[t] = rsqrtf(ideg[t] > 0 ? (float)ideg[t] : 1.0f);
  }
  __syncthreads();

  float m[NPG][2];
#pragma unroll
  for (int n = 0; n < NPG; ++n) { m[n][0] = 0.f; m[n][1] = 0.f; }

  const int base = g * NPG;
#pragma unroll
  for (int e = 0; e < E_PER; ++e) {
    const int s = ls[e], d = ld[e];
    const float w = onorm[s];
    float v0, v1;
    if (FIRST) {
      const float* row = emb + (size_t)feats[base + s] * RANK;
      v0 = row[t]; v1 = row[t + 256];
    } else {
      const unsigned short* row = h_prev + (size_t)(base + s) * RANK;
      v0 = bf16_f32(row[t]); v1 = bf16_f32(row[t + 256]);
    }
    m[d][0] += v0 * w;
    m[d][1] += v1 * w;
  }

#pragma unroll
  for (int n = 0; n < NPG; ++n) {
    const float w = inorm[n];
    const size_t o = (size_t)(base + n) * RANK;
    a_out[o + t]       = bf16_bits(m[n][0] * w);
    a_out[o + t + 256] = bf16_bits(m[n][1] * w);
  }
}

// ---------------------------------------------------------------------------
// GEMM [64 x 512] = A[64 x 512] @ W[512 x 512]  (bf16 in, f32 acc) fused with
// bias + LayerNorm(feature dim) + ReLU, bf16 out.
// Block = 256 threads = 8 waves. Wave w owns columns [w*64, w*64+64)
// (4 N-tiles); 4 M-tiles; K loop 16 x 32.
// A tile staged into LDS by the Tensor Data Mover (one tensor_load_to_lds per
// block), using D# padding (4 DWORDs per 1024B row) to produce the
// bank-conflict-free 520-halfword LDS row stride.
// ---------------------------------------------------------------------------
#define A_STRIDE 520   // 512 + 8 halfwords padding -> 1040B row stride, 16B aligned

__global__ __launch_bounds__(256)
void k_gemm_ln(const unsigned short* __restrict__ A,    // [NNODES][512] bf16
               const unsigned short* __restrict__ WT,   // [512 n][512 k] bf16
               const float* __restrict__ bias,
               const float* __restrict__ gamma,
               const float* __restrict__ beta,
               unsigned short* __restrict__ H) {
  __shared__ unsigned short a_lds[64 * A_STRIDE];
  __shared__ float rs_sum[64];
  __shared__ float rs_sq[64];

  const int t    = threadIdx.x;
  const int wave = t >> 5;
  const int lane = t & 31;
  const int l15  = lane & 15;
  const int hi   = lane >> 4;          // half-wave selector
  const int row_base = blockIdx.x * 64;
  const int colbase  = wave * 64;

  if (t < 64) { rs_sum[t] = 0.f; rs_sq[t] = 0.f; }

  // ---- TDM: stage A tile (64 rows x 1024B) into padded LDS ----
  if (wave == 0) {
    const unsigned lds_off = (unsigned)(size_t)(&a_lds[0]);         // low 32b of flat = LDS offset
    const unsigned long long ga =
        (unsigned long long)(size_t)A + (unsigned long long)row_base * (RANK * 2);
    // D# group 0: count=1, lds_addr, global_addr(57b), type=2
    v4u g0;
    g0.x = 1u;                               // count=1, user mode
    g0.y = lds_off;                          // lds_addr
    g0.z = (unsigned)(ga & 0xffffffffu);     // global_addr[31:0]
    g0.w = (unsigned)((ga >> 32) & 0x1ffffffu) | (2u << 30);  // addr[56:32] | type=2
    // D# group 1: data_size=3 (8B), pad_enable, pad_interval=7 (256 DW = 1024B),
    // pad_amount=3 (4 DW = 16B); tensor/tile dim0 = 128 (8B units), dim1 = 64 rows;
    // tensor_dim0_stride = 128 (contiguous rows).
    v8i g1;
    g1[0] = (int)((3u << 16) | (1u << 20) | (7u << 22) | (3u << 25));
    g1[1] = (int)(128u << 16);   // tensor_dim0[15:0] in bits 63:48
    g1[2] = (int)(64u << 16);    // tensor_dim1[15:0] in bits 95:80
    g1[3] = (int)(128u << 16);   // tile_dim0 in bits 127:112
    g1[4] = 64;                  // tile_dim1 in bits 143:128
    g1[5] = 128;                 // tensor_dim0_stride[31:0]
    g1[6] = 0;
    g1[7] = 0;
    v4i gz4 = {0, 0, 0, 0};
    v8i gz8 = {0, 0, 0, 0, 0, 0, 0, 0};
    __builtin_amdgcn_tensor_load_to_lds(g0, g1, gz4, gz4, gz8, 0);
    __builtin_amdgcn_s_wait_tensorcnt(0);
  }
  __syncthreads();

  v8f acc[4][4];
#pragma unroll
  for (int mt = 0; mt < 4; ++mt)
#pragma unroll
    for (int nt = 0; nt < 4; ++nt) acc[mt][nt] = (v8f)(0.f);

  for (int kc = 0; kc < 16; ++kc) {
    const int k0 = kc * 32;
    // A fragments (16x32 bf16): lane<16 holds K {0..7,16..23}, lane>=16 {8..15,24..31}
    v16bf afr[4];
#pragma unroll
    for (int mt = 0; mt < 4; ++mt) {
      const unsigned short* p = &a_lds[(mt * 16 + l15) * A_STRIDE + k0 + hi * 8];
      uint4 q0 = *(const uint4*)p;
      uint4 q1 = *(const uint4*)(p + 16);
      afr[mt] = make_frag(q0, q1);
    }
#pragma unroll
    for (int nt = 0; nt < 4; ++nt) {
      // B fragment (32x16 bf16): lane<16 holds K {0..7,16..23} at col=lane,
      // lane>=16 holds K {8..15,24..31} at col=lane-16. WT rows are contiguous K.
      const unsigned short* bp =
          WT + (size_t)(colbase + nt * 16 + l15) * RANK + k0 + hi * 8;
      uint4 q0 = *(const uint4*)bp;
      uint4 q1 = *(const uint4*)(bp + 16);
      v16bf bfr = make_frag(q0, q1);
#pragma unroll
      for (int mt = 0; mt < 4; ++mt) {
        acc[mt][nt] = __builtin_amdgcn_wmma_f32_16x16x32_bf16(
            false, afr[mt], false, bfr, (short)0, acc[mt][nt], false, false);
      }
    }
  }

  // Epilogue: bias, row stats (LDS float atomics), LN, ReLU, bf16 store.
  float bn[4], gn[4], btn[4];
#pragma unroll
  for (int nt = 0; nt < 4; ++nt) {
    const int c = colbase + nt * 16 + l15;
    bn[nt] = bias[c]; gn[nt] = gamma[c]; btn[nt] = beta[c];
  }
#pragma unroll
  for (int mt = 0; mt < 4; ++mt) {
#pragma unroll
    for (int j = 0; j < 8; ++j) {
      float s = 0.f, s2 = 0.f;
#pragma unroll
      for (int nt = 0; nt < 4; ++nt) {
        float x = acc[mt][nt][j] + bn[nt];
        acc[mt][nt][j] = x;
        s += x; s2 += x * x;
      }
      const int r = mt * 16 + hi * 8 + j;   // C/D layout: M = j + 8*(lane>>4)
      atomicAdd(&rs_sum[r], s);
      atomicAdd(&rs_sq[r], s2);
    }
  }
  __syncthreads();

#pragma unroll
  for (int mt = 0; mt < 4; ++mt) {
#pragma unroll
    for (int j = 0; j < 8; ++j) {
      const int r = mt * 16 + hi * 8 + j;
      const float mu   = rs_sum[r] * (1.f / 512.f);
      const float var  = rs_sq[r] * (1.f / 512.f) - mu * mu;
      const float rstd = rsqrtf(var + LN_EPS);
      const size_t gro = (size_t)(row_base + r) * RANK;
#pragma unroll
      for (int nt = 0; nt < 4; ++nt) {
        float x = (acc[mt][nt][j] - mu) * rstd * gn[nt] + btn[nt];
        x = fmaxf(x, 0.f);
        H[gro + colbase + nt * 16 + l15] = bf16_bits(x);
      }
    }
  }
}

// ---------------------------------------------------------------------------
// Readout: out[g] = dot(h[g*NPG + 0], w_out) + b_out. One wave per graph.
// ---------------------------------------------------------------------------
__global__ void k_out(const unsigned short* __restrict__ H,
                      const float* __restrict__ w_out,
                      const float* __restrict__ b_out,
                      float* __restrict__ out) {
  const int wid  = threadIdx.x >> 5;
  const int lane = threadIdx.x & 31;
  const int g    = blockIdx.x * 8 + wid;
  const unsigned short* row = H + (size_t)g * NPG * RANK;  // node 0 of graph g
  float s = 0.f;
#pragma unroll
  for (int f = lane; f < RANK; f += 32) s += bf16_f32(row[f]) * w_out[f];
#pragma unroll
  for (int off = 16; off > 0; off >>= 1) s += __shfl_xor(s, off, 32);
  if (lane == 0) out[g] = s + b_out[0];
}

// ---------------------------------------------------------------------------
extern "C" void kernel_launch(void* const* d_in, const int* in_sizes, int n_in,
                              void* d_out, int out_size, void* d_ws, size_t ws_size,
                              hipStream_t stream) {
  (void)in_sizes; (void)n_in; (void)out_size; (void)ws_size;
  const int*   feats  = (const int*)d_in[0];
  const int*   src    = (const int*)d_in[1];
  const int*   dst    = (const int*)d_in[2];
  const float* emb    = (const float*)d_in[3];
  const float* Ws     = (const float*)d_in[4];
  const float* bs     = (const float*)d_in[5];
  const float* gammas = (const float*)d_in[6];
  const float* betas  = (const float*)d_in[7];
  const float* w_out  = (const float*)d_in[8];
  const float* b_out  = (const float*)d_in[9];

  // workspace layout (bf16 halfwords)
  unsigned short* WT   = (unsigned short*)d_ws;             // 3 * 512 * 512
  unsigned short* Abuf = WT + (size_t)3 * 262144;           // NNODES * 512
  unsigned short* Hbuf = Abuf + (size_t)NNODES * RANK;      // NNODES * 512

  k_convert_w<<<3 * 262144 / 256, 256, 0, stream>>>(Ws, WT);

  // layer 0
  k_agg<true><<<NGRAPH, 256, 0, stream>>>(feats, src, dst, emb, nullptr, Abuf);
  k_gemm_ln<<<NNODES / 64, 256, 0, stream>>>(Abuf, WT, bs, gammas, betas, Hbuf);
  // layer 1
  k_agg<false><<<NGRAPH, 256, 0, stream>>>(feats, src, dst, emb, Hbuf, Abuf);
  k_gemm_ln<<<NNODES / 64, 256, 0, stream>>>(Abuf, WT + 262144, bs + 512,
                                             gammas + 512, betas + 512, Hbuf);
  // layer 2
  k_agg<false><<<NGRAPH, 256, 0, stream>>>(feats, src, dst, emb, Hbuf, Abuf);
  k_gemm_ln<<<NNODES / 64, 256, 0, stream>>>(Abuf, WT + 2 * 262144, bs + 1024,
                                             gammas + 1024, betas + 1024, Hbuf);

  k_out<<<NGRAPH / 8, 256, 0, stream>>>(Hbuf, w_out, b_out, (float*)d_out);
}
